// EdgeNetVAE_8177617731796
// MI455X (gfx1250) — matrix operand order repaired and checked by reading, not compiled, not run on hardware
//
#include <hip/hip_runtime.h>
#include <hip/hip_bf16.h>

#define N_NODES 100000
#define N_EDGES 3200000
#define WPB 8
#define BLOCK 256
#define BN_EPS 1e-5f

typedef __attribute__((ext_vector_type(16))) _Float16 v16h;
typedef __attribute__((ext_vector_type(8)))  float    v8f;

union HFrag { v16h v; float4 q[2]; };

__device__ __forceinline__ void lds_fence() {
    asm volatile("s_wait_dscnt 0x0" ::: "memory");
}

__device__ __forceinline__ void atomAddF(float* p, float v) {
    __hip_atomic_fetch_add(p, v, __ATOMIC_RELAXED, __HIP_MEMORY_SCOPE_AGENT);
}

// ---------------- BatchNorm statistics (sum, sumsq per dim) ----------------
__global__ void bn_stats_kernel(const float* __restrict__ x, float* __restrict__ stats) {
    float s0=0,s1=0,s2=0,s3=0,q0=0,q1=0,q2=0,q3=0;
    int stride = gridDim.x * blockDim.x;
    for (int i = blockIdx.x * blockDim.x + threadIdx.x; i < N_NODES; i += stride) {
        float4 v = ((const float4*)x)[i];
        s0 += v.x; s1 += v.y; s2 += v.z; s3 += v.w;
        q0 += v.x*v.x; q1 += v.y*v.y; q2 += v.z*v.z; q3 += v.w*v.w;
    }
#pragma unroll
    for (int m = 16; m >= 1; m >>= 1) {
        s0 += __shfl_xor(s0, m, 32); s1 += __shfl_xor(s1, m, 32);
        s2 += __shfl_xor(s2, m, 32); s3 += __shfl_xor(s3, m, 32);
        q0 += __shfl_xor(q0, m, 32); q1 += __shfl_xor(q1, m, 32);
        q2 += __shfl_xor(q2, m, 32); q3 += __shfl_xor(q3, m, 32);
    }
    if ((threadIdx.x & 31) == 0) {
        atomAddF(&stats[0], s0); atomAddF(&stats[1], s1);
        atomAddF(&stats[2], s2); atomAddF(&stats[3], s3);
        atomAddF(&stats[4], q0); atomAddF(&stats[5], q1);
        atomAddF(&stats[6], q2); atomAddF(&stats[7], q3);
    }
}

__global__ void bn_apply_kernel(const float* __restrict__ x, const float* __restrict__ stats,
                                const float* __restrict__ gamma, const float* __restrict__ beta,
                                float* __restrict__ xn) {
    int i = blockIdx.x * blockDim.x + threadIdx.x;
    if (i >= N_NODES) return;
    const float invN = 1.0f / (float)N_NODES;
    float4 v = ((const float4*)x)[i];
    float m0 = stats[0]*invN, m1 = stats[1]*invN, m2 = stats[2]*invN, m3 = stats[3]*invN;
    float s0 = rsqrtf(stats[4]*invN - m0*m0 + BN_EPS) * gamma[0];
    float s1 = rsqrtf(stats[5]*invN - m1*m1 + BN_EPS) * gamma[1];
    float s2 = rsqrtf(stats[6]*invN - m2*m2 + BN_EPS) * gamma[2];
    float s3 = rsqrtf(stats[7]*invN - m3*m3 + BN_EPS) * gamma[3];
    float4 r;
    r.x = (v.x - m0) * s0 + beta[0];
    r.y = (v.y - m1) * s1 + beta[1];
    r.z = (v.z - m2) * s2 + beta[2];
    r.w = (v.w - m3) * s3 + beta[3];
    ((float4*)xn)[i] = r;
}

// ---------------- Encoder EdgeConv: feat8 -> 32 (VALU) -> 32 (WMMA) -> scatter ----------------
__global__ __launch_bounds__(BLOCK)
void enc_edge_kernel(const int* __restrict__ src, const int* __restrict__ tgt,
                     const float* __restrict__ xn,
                     const float* __restrict__ eW1, const float* __restrict__ eb1,
                     const float* __restrict__ eW2, const float* __restrict__ eb2,
                     float* __restrict__ hsum, float* __restrict__ cnt) {
    __shared__ float sW1[8][32];
    __shared__ float sB1[32];
    __shared__ _Float16 sW2T[32][32] __attribute__((aligned(16)));  // [outCol][k]
    __shared__ float sB2[32];
    __shared__ _Float16 sL1[WPB][16][32] __attribute__((aligned(16)));
    __shared__ int sTgt[WPB][16];

    int tid = threadIdx.x;
    sW1[tid >> 5][tid & 31] = eW1[tid];              // 8*32 = 256
    if (tid < 32) { sB1[tid] = eb1[tid]; sB2[tid] = eb2[tid]; }
#pragma unroll
    for (int i = 0; i < 4; i++) {                    // eW2 [k][c] -> sW2T[c][k] f16
        int idx = tid + i * 256;
        sW2T[idx & 31][idx >> 5] = (_Float16)eW2[idx];
    }
    __syncthreads();

    int lane = tid & 31;
    int w = tid >> 5;

    // Loop-invariant B fragments (32x16 f16, lane=column, K-half by lane group)
    HFrag b0, b1;
    {
        int cN = lane & 15;
        int kb = (lane < 16) ? 0 : 16;
        b0.q[0] = *(const float4*)&sW2T[cN][kb];
        b0.q[1] = *(const float4*)&sW2T[cN][kb + 8];
        b1.q[0] = *(const float4*)&sW2T[16 + cN][kb];
        b1.q[1] = *(const float4*)&sW2T[16 + cN][kb + 8];
    }

    const int nTiles = N_EDGES >> 4;
    int wid = blockIdx.x * WPB + w;
    int nW = gridDim.x * WPB;
    int e  = lane & 15;
    int hb = (lane < 16) ? 0 : 16;   // hidden-column half this lane computes for layer1

    for (int tile = wid; tile < nTiles; tile += nW) {
        lds_fence();
        int ge = tile * 16 + e;
        int sI = src[ge], tI = tgt[ge];
        float4 xi = *(const float4*)&xn[tI * 4];
        float4 xj = *(const float4*)&xn[sI * 4];
        float f[8] = { xi.x, xi.y, xi.z, xi.w,
                       xj.x - xi.x, xj.y - xi.y, xj.z - xi.z, xj.w - xi.w };
        if (lane < 16) {
            sTgt[w][e] = tI;
            atomAddF(&cnt[tI], 1.0f);
        }
        // Layer 1: 16 hidden outputs per lane, K=8, VALU
        union { _Float16 h[16]; float4 q[2]; } t16;
#pragma unroll
        for (int hh = 0; hh < 16; hh++) {
            float acc = sB1[hb + hh];
#pragma unroll
            for (int k = 0; k < 8; k++) acc = fmaf(f[k], sW1[k][hb + hh], acc);
            t16.h[hh] = (_Float16)fmaxf(acc, 0.0f);
        }
        {
            float4* dst = (float4*)&sL1[w][e][hb];
            dst[0] = t16.q[0];
            dst[1] = t16.q[1];
        }
        lds_fence();
        // A fragment: 16x32 f16 layout
        HFrag a;
        {
            int m  = lane & 15;
            int kb = (lane < 16) ? 0 : 16;
            a.q[0] = *(const float4*)&sL1[w][m][kb];
            a.q[1] = *(const float4*)&sL1[w][m][kb + 8];
        }
        v8f c0 = {}, c1 = {};
        c0 = __builtin_amdgcn_wmma_f32_16x16x32_f16(false, a.v, false, b0.v, (short)0, c0, false, false);
        c1 = __builtin_amdgcn_wmma_f32_16x16x32_f16(false, a.v, false, b1.v, (short)0, c1, false, false);
        // Epilogue: bias + relu + scatter (C layout: VGPR r -> row r / 8+r)
        int n  = lane & 15;
        int rb = (lane < 16) ? 0 : 8;
#pragma unroll
        for (int r = 0; r < 8; r++) {
            int row = rb + r;
            int tg = sTgt[w][row];
            float v0 = fmaxf(c0[r] + sB2[n], 0.0f);
            float v1 = fmaxf(c1[r] + sB2[16 + n], 0.0f);
            atomAddF(&hsum[tg * 32 + n], v0);
            atomAddF(&hsum[tg * 32 + 16 + n], v1);
        }
    }
}

// ---------------- Nodes: mean, mu/logvar heads, reparameterize ----------------
__global__ void node_mid_kernel(const float* __restrict__ hsum, const float* __restrict__ cnt,
                                const float* __restrict__ mW, const float* __restrict__ mb,
                                const float* __restrict__ vW, const float* __restrict__ vb,
                                const float* __restrict__ eps, float* __restrict__ out,
                                float* __restrict__ z) {
    int i = blockIdx.x * blockDim.x + threadIdx.x;
    if (i >= N_NODES) return;
    float inv = 1.0f / fmaxf(cnt[i], 1.0f);
    float mu0 = mb[0], mu1 = mb[1], lv0 = vb[0], lv1 = vb[1];
    const float4* hp = (const float4*)&hsum[i * 32];
#pragma unroll
    for (int k4 = 0; k4 < 8; k4++) {
        float4 hv = hp[k4];
        float hl[4] = { hv.x, hv.y, hv.z, hv.w };
#pragma unroll
        for (int j = 0; j < 4; j++) {
            int k = k4 * 4 + j;
            float hk = hl[j] * inv;
            mu0 = fmaf(hk, mW[k * 2 + 0], mu0);
            mu1 = fmaf(hk, mW[k * 2 + 1], mu1);
            lv0 = fmaf(hk, vW[k * 2 + 0], lv0);
            lv1 = fmaf(hk, vW[k * 2 + 1], lv1);
        }
    }
    float z0 = mu0 + eps[i * 2 + 0] * __expf(0.5f * lv0);
    float z1 = mu1 + eps[i * 2 + 1] * __expf(0.5f * lv1);
    out[N_NODES * 4 + i * 2 + 0] = mu0;
    out[N_NODES * 4 + i * 2 + 1] = mu1;
    out[N_NODES * 6 + i * 2 + 0] = lv0;
    out[N_NODES * 6 + i * 2 + 1] = lv1;
    z[i * 2 + 0] = z0;
    z[i * 2 + 1] = z1;
}

// ---------------- Decoder EdgeConv: feat4 -> 32 (VALU) -> 32 (WMMA) -> 4 (LDS dot) ----------------
__global__ __launch_bounds__(BLOCK)
void dec_edge_kernel(const int* __restrict__ src, const int* __restrict__ tgt,
                     const float* __restrict__ z,
                     const float* __restrict__ dW1, const float* __restrict__ db1,
                     const float* __restrict__ dW2, const float* __restrict__ db2,
                     const float* __restrict__ dW3, const float* __restrict__ db3,
                     float* __restrict__ osum) {
    __shared__ float sW1[4][32];
    __shared__ float sB1[32];
    __shared__ _Float16 sW2T[32][32] __attribute__((aligned(16)));
    __shared__ float sB2[32];
    __shared__ float sW3[32][4];
    __shared__ float sB3[4];
    __shared__ _Float16 sL1[WPB][16][32] __attribute__((aligned(16)));
    __shared__ float sL2[WPB][16][33];
    __shared__ int sTgt[WPB][16];

    int tid = threadIdx.x;
    if (tid < 128) sW1[tid >> 5][tid & 31] = dW1[tid];      // 4*32
    if (tid < 32) { sB1[tid] = db1[tid]; sB2[tid] = db2[tid]; }
    if (tid >= 128 && tid < 256) sW3[(tid - 128) >> 2][tid & 3] = dW3[tid - 128]; // 32*4
    if (tid < 4) sB3[tid] = db3[tid];
#pragma unroll
    for (int i = 0; i < 4; i++) {
        int idx = tid + i * 256;
        sW2T[idx & 31][idx >> 5] = (_Float16)dW2[idx];
    }
    __syncthreads();

    int lane = tid & 31;
    int w = tid >> 5;

    HFrag b0, b1;
    {
        int cN = lane & 15;
        int kb = (lane < 16) ? 0 : 16;
        b0.q[0] = *(const float4*)&sW2T[cN][kb];
        b0.q[1] = *(const float4*)&sW2T[cN][kb + 8];
        b1.q[0] = *(const float4*)&sW2T[16 + cN][kb];
        b1.q[1] = *(const float4*)&sW2T[16 + cN][kb + 8];
    }

    const int nTiles = N_EDGES >> 4;
    int wid = blockIdx.x * WPB + w;
    int nW = gridDim.x * WPB;
    int e  = lane & 15;
    int hb = (lane < 16) ? 0 : 16;

    for (int tile = wid; tile < nTiles; tile += nW) {
        lds_fence();
        int ge = tile * 16 + e;
        int sI = src[ge], tI = tgt[ge];
        float2 zi = *(const float2*)&z[tI * 2];
        float2 zj = *(const float2*)&z[sI * 2];
        float f[4] = { zi.x, zi.y, zj.x - zi.x, zj.y - zi.y };
        if (lane < 16) sTgt[w][e] = tI;

        union { _Float16 h[16]; float4 q[2]; } t16;
#pragma unroll
        for (int hh = 0; hh < 16; hh++) {
            float acc = sB1[hb + hh];
#pragma unroll
            for (int k = 0; k < 4; k++) acc = fmaf(f[k], sW1[k][hb + hh], acc);
            t16.h[hh] = (_Float16)fmaxf(acc, 0.0f);
        }
        {
            float4* dst = (float4*)&sL1[w][e][hb];
            dst[0] = t16.q[0];
            dst[1] = t16.q[1];
        }
        lds_fence();
        HFrag a;
        {
            int m  = lane & 15;
            int kb = (lane < 16) ? 0 : 16;
            a.q[0] = *(const float4*)&sL1[w][m][kb];
            a.q[1] = *(const float4*)&sL1[w][m][kb + 8];
        }
        v8f c0 = {}, c1 = {};
        c0 = __builtin_amdgcn_wmma_f32_16x16x32_f16(false, a.v, false, b0.v, (short)0, c0, false, false);
        c1 = __builtin_amdgcn_wmma_f32_16x16x32_f16(false, a.v, false, b1.v, (short)0, c1, false, false);
        // bias + relu, stage to LDS for the 32->4 layer
        int n  = lane & 15;
        int rb = (lane < 16) ? 0 : 8;
#pragma unroll
        for (int r = 0; r < 8; r++) {
            int row = rb + r;
            sL2[w][row][n]      = fmaxf(c0[r] + sB2[n], 0.0f);
            sL2[w][row][16 + n] = fmaxf(c1[r] + sB2[16 + n], 0.0f);
        }
        lds_fence();
        // Layer 3 (no relu): 64 (edge,out) pairs over 32 lanes
#pragma unroll
        for (int j = 0; j < 2; j++) {
            int p = lane * 2 + j;
            int e3 = p >> 2, o = p & 3;
            float acc = sB3[o];
#pragma unroll
            for (int k = 0; k < 32; k++) acc = fmaf(sL2[w][e3][k], sW3[k][o], acc);
            atomAddF(&osum[sTgt[w][e3] * 4 + o], acc);
        }
    }
}

// ---------------- Finalize: out = osum / max(count,1) ----------------
__global__ void finalize_kernel(const float* __restrict__ osum, const float* __restrict__ cnt,
                                float* __restrict__ out) {
    int i = blockIdx.x * blockDim.x + threadIdx.x;
    if (i >= N_NODES * 4) return;
    out[i] = osum[i] / fmaxf(cnt[i >> 2], 1.0f);
}

extern "C" void kernel_launch(void* const* d_in, const int* in_sizes, int n_in,
                              void* d_out, int out_size, void* d_ws, size_t ws_size,
                              hipStream_t stream) {
    (void)in_sizes; (void)n_in; (void)out_size; (void)ws_size;
    const float* x     = (const float*)d_in[0];
    const int*   ei    = (const int*)d_in[1];
    const float* eps   = (const float*)d_in[2];
    const float* gamma = (const float*)d_in[3];
    const float* beta  = (const float*)d_in[4];
    const float* eW1 = (const float*)d_in[5];
    const float* eb1 = (const float*)d_in[6];
    const float* eW2 = (const float*)d_in[7];
    const float* eb2 = (const float*)d_in[8];
    const float* mW  = (const float*)d_in[9];
    const float* mb  = (const float*)d_in[10];
    const float* vW  = (const float*)d_in[11];
    const float* vb  = (const float*)d_in[12];
    const float* dW1 = (const float*)d_in[13];
    const float* db1 = (const float*)d_in[14];
    const float* dW2 = (const float*)d_in[15];
    const float* db2 = (const float*)d_in[16];
    const float* dW3 = (const float*)d_in[17];
    const float* db3 = (const float*)d_in[18];
    float* out = (float*)d_out;
    float* ws  = (float*)d_ws;

    // Workspace layout (floats)
    float* stats = ws;               // 16
    float* xn    = ws + 16;          // N*4
    float* hsum  = ws + 400016;      // N*32
    float* cnt   = ws + 3600016;     // N
    float* zbuf  = ws + 3700016;     // N*2
    float* osum  = ws + 3900016;     // N*4

    const int* src  = ei;
    const int* tgtp = ei + N_EDGES;

    hipMemsetAsync(stats, 0, 16 * sizeof(float), stream);
    hipMemsetAsync(hsum, 0, (size_t)(N_NODES * 32 + N_NODES) * sizeof(float), stream); // hsum + cnt
    hipMemsetAsync(osum, 0, (size_t)(N_NODES * 4) * sizeof(float), stream);

    bn_stats_kernel<<<256, 256, 0, stream>>>(x, stats);
    bn_apply_kernel<<<(N_NODES + 255) / 256, 256, 0, stream>>>(x, stats, gamma, beta, xn);
    enc_edge_kernel<<<2048, BLOCK, 0, stream>>>(src, tgtp, xn, eW1, eb1, eW2, eb2, hsum, cnt);
    node_mid_kernel<<<(N_NODES + 255) / 256, 256, 0, stream>>>(hsum, cnt, mW, mb, vW, vb, eps, out, zbuf);
    dec_edge_kernel<<<2048, BLOCK, 0, stream>>>(src, tgtp, zbuf, dW1, db1, dW2, db2, dW3, db3, osum);
    finalize_kernel<<<(N_NODES * 4 + 255) / 256, 256, 0, stream>>>(osum, cnt, out);
}
